// MLAAttention_56753697849914
// MI455X (gfx1250) — matrix operand order repaired and checked by reading, not compiled, not run on hardware
//
#include <hip/hip_runtime.h>

#define BB    2
#define SS    2048
#define DIMM  2048
#define NH    16
#define QD    192
#define NOPE  128
#define ROPED 64
#define VD    128
#define KVL   512
#define MTOT  (BB * SS)          // 4096 rows (b*s)
#define NQ    (NH * QD)          // 3072
#define NKVA  (KVL + ROPED)      // 576
#define NKVB  (NH * (NOPE + VD)) // 4096
#define KSTEP 64

typedef unsigned short u16;
typedef __attribute__((ext_vector_type(16))) __bf16       v16bf;
typedef __attribute__((ext_vector_type(8)))  float        v8f;
typedef __attribute__((ext_vector_type(4)))  unsigned int u32x4;
typedef __attribute__((ext_vector_type(4)))  int          v4i;

typedef __attribute__((address_space(1))) v4i* gv4i_p;   // global
typedef __attribute__((address_space(3))) v4i* lv4i_p;   // LDS

union Frag {
  v16bf v;
  u16   u[16];
  u32x4 q[2];
};

__device__ inline u16 f32_to_bf16(float f) {
  unsigned int x = __float_as_uint(f);
  x += 0x7FFFu + ((x >> 16) & 1u);   // round-to-nearest-even
  return (u16)(x >> 16);
}

// A-matrix fragment, 16x32 bf16 (rows row0..row0+15, k = k0..k0+31).
__device__ inline v16bf load_a(const u16* __restrict__ base, int ld,
                               int row0, int k0, int r, int hi) {
  const u16* p = base + (size_t)(row0 + r) * ld + k0 + hi * 8;
  Frag f;
  f.q[0] = *(const u32x4*)p;
  f.q[1] = *(const u32x4*)(p + 16);
  return f.v;
}

// B-matrix fragment, 32x16 bf16 from row-major W[N,K]: column n = W row n.
__device__ inline v16bf load_b(const u16* __restrict__ base, int ld,
                               int n0, int k0, int r, int hi) {
  const u16* p = base + (size_t)(n0 + r) * ld + k0 + hi * 8;
  Frag f;
  f.q[0] = *(const u32x4*)p;
  f.q[1] = *(const u32x4*)(p + 16);
  return f.v;
}

#define WMMA_BF16(A, B, C) \
  __builtin_amdgcn_wmma_f32_16x16x32_bf16(false, (A), false, (B), (short)0, (C), false, false)

// --------------------------------------------------------- async copy helper
#if __has_builtin(__builtin_amdgcn_global_load_async_to_lds_b128)
#define USE_ASYNC_LDS 1
#endif

__device__ inline void copy16B_g2l(u16* lds, const u16* g) {
#ifdef USE_ASYNC_LDS
  __builtin_amdgcn_global_load_async_to_lds_b128(
      (gv4i_p)(v4i*)(void*)g,
      (lv4i_p)(v4i*)(void*)lds, 0, 0);
#else
  *(u32x4*)lds = *(const u32x4*)g;
#endif
}

__device__ inline void async_copy_fence() {
#ifdef USE_ASYNC_LDS
  asm volatile("s_wait_asynccnt 0x0" ::: "memory");
#endif
}

// ---------------------------------------------------------------- convert
__global__ __launch_bounds__(256) void cvt_f32_bf16(const float* __restrict__ in,
                                                    u16* __restrict__ out, size_t n) {
  size_t i = (size_t)blockIdx.x * blockDim.x + threadIdx.x;
  size_t stride = (size_t)gridDim.x * blockDim.x;
  for (; i < n; i += stride) out[i] = f32_to_bf16(in[i]);
}

// ---------------------------------------------------------------- GEMM
// C[M,N](f32) = A[M,K](bf16) * W[N,K](bf16)^T
// block tile 128x64, double-buffered LDS staging (async), K-step 64.
// 8 waves -> wave tile 32x32 -> 2x2 WMMA tiles fed by ds_load_b128.
__global__ __launch_bounds__(256) void gemm_bf16_lds(
    const u16* __restrict__ A, const u16* __restrict__ W, float* __restrict__ C,
    int M, int N, int K) {
  __shared__ u16 shA[2][128 * KSTEP];  // 2 x 16 KB
  __shared__ u16 shB[2][64 * KSTEP];   // 2 x  8 KB
  const int tid = threadIdx.x;
  const int lane = tid & 31;
  const int wave = tid >> 5;
  const int r = lane & 15, hi = lane >> 4;
  const int bm = blockIdx.y * 128;
  const int bn = blockIdx.x * 64;
  const int wm = (wave >> 1) * 32;
  const int wn = (wave & 1) * 32;

  auto issue = [&](int buf, int k0) {
    // A tile: 128x64 = 1024 16B chunks, 4 per thread
    for (int c = 0; c < 4; ++c) {
      int chunk = tid + c * 256;
      int row = chunk >> 3;
      int col = (chunk & 7) * 8;
      copy16B_g2l(&shA[buf][row * KSTEP + col], A + (size_t)(bm + row) * K + k0 + col);
    }
    // B tile: 64x64 = 512 chunks, 2 per thread
    for (int c = 0; c < 2; ++c) {
      int chunk = tid + c * 256;
      int row = chunk >> 3;
      int col = (chunk & 7) * 8;
      copy16B_g2l(&shB[buf][row * KSTEP + col], W + (size_t)(bn + row) * K + k0 + col);
    }
  };

  const int ktiles = K / KSTEP;
  issue(0, 0);
  v8f acc00 = {}, acc01 = {}, acc10 = {}, acc11 = {};
  for (int kt = 0; kt < ktiles; ++kt) {
    const int buf = kt & 1;
    async_copy_fence();
    __syncthreads();
    if (kt + 1 < ktiles) issue(buf ^ 1, (kt + 1) * KSTEP);
    const u16* sA = shA[buf];
    const u16* sB = shB[buf];
    for (int kk = 0; kk < KSTEP; kk += 32) {
      v16bf a0 = load_a(sA, KSTEP, wm, kk, r, hi);
      v16bf a1 = load_a(sA, KSTEP, wm + 16, kk, r, hi);
      v16bf b0 = load_b(sB, KSTEP, wn, kk, r, hi);
      v16bf b1 = load_b(sB, KSTEP, wn + 16, kk, r, hi);
      acc00 = WMMA_BF16(a0, b0, acc00);
      acc01 = WMMA_BF16(a0, b1, acc01);
      acc10 = WMMA_BF16(a1, b0, acc10);
      acc11 = WMMA_BF16(a1, b1, acc11);
    }
    __syncthreads();
  }
  for (int v = 0; v < 8; ++v) {
    int row0 = bm + wm + hi * 8 + v;
    int row1 = row0 + 16;
    C[(size_t)row0 * N + bn + wn + r]      = acc00[v];
    C[(size_t)row0 * N + bn + wn + 16 + r] = acc01[v];
    C[(size_t)row1 * N + bn + wn + r]      = acc10[v];
    C[(size_t)row1 * N + bn + wn + 16 + r] = acc11[v];
  }
}

// ---------------------------------------------------------------- RMS norm
__global__ __launch_bounds__(256) void rmsnorm_rows(const float* __restrict__ ckv,
                                                    const float* __restrict__ w,
                                                    u16* __restrict__ out) {
  const int lane = threadIdx.x & 31;
  const int wave = threadIdx.x >> 5;
  const int row = blockIdx.x * 8 + wave;
  const float* p = ckv + (size_t)row * NKVA;
  float vals[16];
  float ss = 0.f;
  for (int i = 0; i < 16; ++i) { float t = p[lane + i * 32]; vals[i] = t; ss += t * t; }
  for (int off = 16; off >= 1; off >>= 1) ss += __shfl_xor(ss, off, 32);
  float inv = rsqrtf(ss * (1.0f / KVL) + 1e-6f);
  u16* o = out + (size_t)row * KVL;
  for (int i = 0; i < 16; ++i) { int c = lane + i * 32; o[c] = f32_to_bf16(vals[i] * inv * w[c]); }
}

// ---------------------------------------------------------------- RoPE q -> [b,h,s,192]
__global__ __launch_bounds__(256) void build_q(const float* __restrict__ q,
                                               const float* __restrict__ rope,
                                               u16* __restrict__ qf) {
  int i = blockIdx.x * 256 + threadIdx.x;   // i = (b*NH + h)*SS + s
  if (i >= BB * NH * SS) return;
  int s = i & (SS - 1);
  int h = (i >> 11) & (NH - 1);
  int b = i >> 15;
  int bs = b * SS + s;
  const float* src = q + (size_t)bs * NQ + h * QD;
  u16* dst = qf + (size_t)i * QD;
  for (int d = 0; d < NOPE; ++d) dst[d] = f32_to_bf16(src[d]);
  const float* cs = rope + (size_t)s * (2 * ROPED);
  for (int j = 0; j < ROPED; ++j) {
    float t = src[NOPE + j];
    float rot = (j < ROPED / 2) ? -src[NOPE + j + ROPED / 2] : src[NOPE + j - ROPED / 2];
    dst[NOPE + j] = f32_to_bf16(t * cs[j] + rot * cs[ROPED + j]);
  }
}

// ---------------------------------------------------------------- RoPE k -> [b,h,s,192]
__global__ __launch_bounds__(256) void build_k(const float* __restrict__ kv,
                                               const float* __restrict__ ckv,
                                               const float* __restrict__ rope,
                                               u16* __restrict__ kf) {
  int i = blockIdx.x * 256 + threadIdx.x;   // i = (b*NH + h)*SS + s
  if (i >= BB * NH * SS) return;
  int s = i & (SS - 1);
  int h = (i >> 11) & (NH - 1);
  int b = i >> 15;
  int bs = b * SS + s;
  const float* kvp = kv + ((size_t)bs * NH + h) * (NOPE + VD);
  u16* dst = kf + (size_t)i * QD;
  for (int d = 0; d < NOPE; ++d) dst[d] = f32_to_bf16(kvp[d]);
  const float* kr = ckv + (size_t)bs * NKVA + KVL;
  const float* cs = rope + (size_t)s * (2 * ROPED);
  for (int j = 0; j < ROPED; ++j) {
    float t = kr[j];
    float rot = (j < ROPED / 2) ? -kr[j + ROPED / 2] : kr[j - ROPED / 2];
    dst[NOPE + j] = f32_to_bf16(t * cs[j] + rot * cs[ROPED + j]);
  }
}

// ---------------------------------------------------------------- V^T -> [b,h,d,s]
// block = (b, h, 64-key tile); coalesced writes along s.
__global__ __launch_bounds__(256) void build_vT(const float* __restrict__ kv,
                                                u16* __restrict__ vT) {
  int blk = blockIdx.x;
  int stile = blk & (SS / 64 - 1);
  int h = (blk >> 5) & (NH - 1);
  int b = blk >> 9;
  int s = stile * 64 + (threadIdx.x & 63);
  int d0 = (threadIdx.x >> 6) * 32;
  const float* src = kv + ((size_t)(b * SS + s) * NH + h) * (NOPE + VD) + NOPE + d0;
  u16* dst = vT + ((size_t)(b * NH + h) * VD + d0) * SS + s;
  for (int j = 0; j < 32; ++j) dst[(size_t)j * SS] = f32_to_bf16(src[j]);
}

// ---------------------------------------------------------------- flash attn
// one wave = one (b, h, 16-row q tile); online softmax; causal.
// q/k layout [b,h,s,192]; v layout [b,h,d,s] (pre-transposed).
__global__ __launch_bounds__(256) void mla_flash(const u16* __restrict__ qf,
                                                 const u16* __restrict__ kf,
                                                 const u16* __restrict__ vT,
                                                 u16* __restrict__ attn) {
  __shared__ u16 pshm[8][16 * 32];
  const int lane = threadIdx.x & 31;
  const int wave = threadIdx.x >> 5;
  const int r = lane & 15, hi = lane >> 4;
  const int widx = blockIdx.x * 8 + wave;
  const int qtile = widx & (SS / 16 - 1);
  const int h = (widx >> 7) & (NH - 1);
  const int b = widx >> 11;
  const int q0 = qtile * 16;

  const u16* Qb = qf + (size_t)(b * NH + h) * SS * QD;
  const u16* Kb = kf + (size_t)(b * NH + h) * SS * QD;
  const u16* Vt = vT + (size_t)(b * NH + h) * VD * SS;

  v16bf qa[6];
  for (int d = 0; d < 6; ++d) qa[d] = load_a(Qb, QD, q0, d * 32, r, hi);

  v8f accO[8];
  for (int n = 0; n < 8; ++n) { v8f z = {}; accO[n] = z; }
  float mrow[8], lrow[8];
  for (int v = 0; v < 8; ++v) { mrow[v] = -1e30f; lrow[v] = 0.f; }
  const float scale = 0.07216878364870323f;  // 192^-0.5
  const int nch = (q0 + 47) >> 5;

  for (int ch = 0; ch < nch; ++ch) {
    const int kk = ch * 32;
    // prefetch next K chunk rows
    if (ch + 1 < nch) {
      int prow = kk + 32 + lane;
      if (prow > SS - 1) prow = SS - 1;
      __builtin_prefetch(Kb + (size_t)prow * QD, 0, 1);
    }
    v8f s0 = {}, s1 = {};
    for (int d = 0; d < 6; ++d) {
      v16bf kb0 = load_b(Kb, QD, kk, d * 32, r, hi);
      s0 = WMMA_BF16(qa[d], kb0, s0);
    }
    for (int d = 0; d < 6; ++d) {
      v16bf kb1 = load_b(Kb, QD, kk + 16, d * 32, r, hi);
      s1 = WMMA_BF16(qa[d], kb1, s1);
    }
    const bool need_mask = (kk + 31) > q0;
    float p0[8], p1[8];
    for (int v = 0; v < 8; ++v) {
      float a = s0[v] * scale, c = s1[v] * scale;
      if (need_mask) {
        int row = q0 + hi * 8 + v;
        if (kk + r > row)      a = -1e30f;
        if (kk + 16 + r > row) c = -1e30f;
      }
      float mx = fmaxf(a, c);
      for (int off = 8; off >= 1; off >>= 1) mx = fmaxf(mx, __shfl_xor(mx, off, 32));
      float mnew = fmaxf(mrow[v], mx);
      float corr = __expf(mrow[v] - mnew);
      a = __expf(a - mnew);
      c = __expf(c - mnew);
      lrow[v] = lrow[v] * corr + a + c;
      mrow[v] = mnew;
      p0[v] = a; p1[v] = c;
      for (int n = 0; n < 8; ++n) accO[n][v] *= corr;
    }
    // C-layout -> A-layout transpose through per-wave LDS slab
    for (int v = 0; v < 8; ++v) {
      pshm[wave][(hi * 8 + v) * 32 + r]      = f32_to_bf16(p0[v]);
      pshm[wave][(hi * 8 + v) * 32 + 16 + r] = f32_to_bf16(p1[v]);
    }
    __builtin_amdgcn_wave_barrier();
    asm volatile("s_wait_dscnt 0x0" ::: "memory");
    Frag pa;
    {
      const u16* pp = &pshm[wave][r * 32 + hi * 8];
      pa.q[0] = *(const u32x4*)pp;
      pa.q[1] = *(const u32x4*)(pp + 16);
    }
    __builtin_amdgcn_wave_barrier();
    // P(16x32) @ V(32x128) via V^T: vector B-fragment loads
    for (int n = 0; n < 8; ++n) {
      v16bf vb = load_b(Vt, SS, n * 16, kk, r, hi);
      accO[n] = WMMA_BF16(pa.v, vb, accO[n]);
    }
  }
  for (int v = 0; v < 8; ++v) {
    float l = lrow[v];
    for (int off = 8; off >= 1; off >>= 1) l += __shfl_xor(l, off, 32);
    float inv = 1.f / l;
    int row = q0 + hi * 8 + v;
    u16* op = attn + ((size_t)(b * SS + row)) * (NH * VD) + h * VD;
    for (int n = 0; n < 8; ++n) op[n * 16 + r] = f32_to_bf16(accO[n][v] * inv);
  }
}

// ---------------------------------------------------------------- launch
extern "C" void kernel_launch(void* const* d_in, const int* in_sizes, int n_in,
                              void* d_out, int out_size, void* d_ws, size_t ws_size,
                              hipStream_t stream) {
  const float* x     = (const float*)d_in[0];
  const float* rope  = (const float*)d_in[1];
  const float* wq    = (const float*)d_in[2];
  const float* wkva  = (const float*)d_in[3];
  const float* normw = (const float*)d_in[4];
  const float* wkvb  = (const float*)d_in[5];
  const float* wo    = (const float*)d_in[6];
  float* out = (float*)d_out;

  char* ws = (char*)d_ws;
  size_t off = 0;
  auto alloc = [&](size_t bytes) -> char* {
    char* p = ws + off;
    off += (bytes + 255) & ~(size_t)255;
    return p;
  };
  u16*   xb    = (u16*)alloc((size_t)MTOT * DIMM * 2);
  u16*   wqb   = (u16*)alloc((size_t)NQ * DIMM * 2);
  u16*   wkvab = (u16*)alloc((size_t)NKVA * DIMM * 2);
  u16*   wkvbb = (u16*)alloc((size_t)NKVB * KVL * 2);
  u16*   wob   = (u16*)alloc((size_t)DIMM * DIMM * 2);
  float* qf32  = (float*)alloc((size_t)MTOT * NQ * 4);
  float* ckv   = (float*)alloc((size_t)MTOT * NKVA * 4);
  u16*   ckvn  = (u16*)alloc((size_t)MTOT * KVL * 2);
  float* kv32  = (float*)alloc((size_t)MTOT * NKVB * 4);
  u16*   qfb   = (u16*)alloc((size_t)MTOT * NH * QD * 2);
  u16*   kfb   = (u16*)alloc((size_t)MTOT * NH * QD * 2);
  u16*   vTb   = (u16*)alloc((size_t)MTOT * NH * VD * 2);
  u16*   attnb = (u16*)alloc((size_t)MTOT * NH * VD * 2);

  auto cvt = [&](const float* in, u16* o, size_t n) {
    int blocks = (int)((n + 255) / 256);
    if (blocks > 4096) blocks = 4096;
    cvt_f32_bf16<<<blocks, 256, 0, stream>>>(in, o, n);
  };
  cvt(x, xb, (size_t)MTOT * DIMM);
  cvt(wq, wqb, (size_t)NQ * DIMM);
  cvt(wkva, wkvab, (size_t)NKVA * DIMM);
  cvt(wkvb, wkvbb, (size_t)NKVB * KVL);
  cvt(wo, wob, (size_t)DIMM * DIMM);

  gemm_bf16_lds<<<dim3(NQ / 64, MTOT / 128), 256, 0, stream>>>(xb, wqb, qf32, MTOT, NQ, DIMM);
  gemm_bf16_lds<<<dim3(NKVA / 64, MTOT / 128), 256, 0, stream>>>(xb, wkvab, ckv, MTOT, NKVA, DIMM);
  rmsnorm_rows<<<MTOT / 8, 256, 0, stream>>>(ckv, normw, ckvn);
  gemm_bf16_lds<<<dim3(NKVB / 64, MTOT / 128), 256, 0, stream>>>(ckvn, wkvbb, kv32, MTOT, NKVB, KVL);
  build_q<<<(BB * NH * SS) / 256, 256, 0, stream>>>(qf32, rope, qfb);
  build_k<<<(BB * NH * SS) / 256, 256, 0, stream>>>(kv32, ckv, rope, kfb);
  build_vT<<<BB * NH * (SS / 64), 256, 0, stream>>>(kv32, vTb);
  mla_flash<<<(BB * NH * (SS / 16)) / 8, 256, 0, stream>>>(qfb, kfb, vTb, attnb);
  gemm_bf16_lds<<<dim3(DIMM / 64, MTOT / 128), 256, 0, stream>>>(attnb, wob, out, MTOT, DIMM, DIMM);
}
